// AttentionPoolingTemporalEncoder_13469017440427
// MI455X (gfx1250) — matrix operand order repaired and checked by reading, not compiled, not run on hardware
//
#include <hip/hip_runtime.h>

typedef __attribute__((ext_vector_type(16))) __bf16 v16bf;
typedef __attribute__((ext_vector_type(8)))  __bf16 v8bf;
typedef __attribute__((ext_vector_type(4)))  __bf16 v4bf;
typedef __attribute__((ext_vector_type(8)))  float  v8f;

#define B_  32
#define S_  4096
#define IN_ 1024
#define E_  512
#define H_  8
#define D_  64
#define LDA 48   // padded LDS row stride in halves (96B, keeps b128 alignment)

static __device__ __forceinline__ __bf16 f2bf(float f) {
  unsigned int u = __builtin_bit_cast(unsigned int, f);
  u += 0x7FFFu + ((u >> 16) & 1u);                  // round-to-nearest-even
  unsigned short s = (unsigned short)(u >> 16);
  return __builtin_bit_cast(__bf16, s);
}
static __device__ __forceinline__ float bf2f(__bf16 b) {
  unsigned int u = ((unsigned int)__builtin_bit_cast(unsigned short, b)) << 16;
  return __builtin_bit_cast(float, u);
}

// ---------------------------------------------------------------------------
// k0: qh = query@Wq + bq ; qk[h][e] = sum_d qh[h,d]*Wk[e,h*64+d] ; qb[h]
// ---------------------------------------------------------------------------
__global__ __launch_bounds__(256) void k0_query(
    const float* __restrict__ query, const float* __restrict__ Wq,
    const float* __restrict__ bq,    const float* __restrict__ Wk,
    const float* __restrict__ bk,    float* __restrict__ qk,
    float* __restrict__ qb)
{
  __shared__ float qhL[E_];
  const int tid = (int)threadIdx.x;
  for (int o = tid; o < E_; o += 256) {
    float sv = bq[o];
    for (int i = 0; i < E_; ++i) sv += query[i] * Wq[(size_t)i * E_ + o];
    qhL[o] = sv;
  }
  __syncthreads();
  for (int idx = tid; idx < H_ * E_; idx += 256) {
    const int hh = idx >> 9, e = idx & (E_ - 1);
    float sv = 0.f;
    for (int d = 0; d < D_; ++d) sv += qhL[hh * D_ + d] * Wk[(size_t)e * E_ + hh * D_ + d];
    qk[idx] = sv;                                  // layout [h][e]
  }
  if (tid < H_) {
    float sv = 0.f;
    for (int d = 0; d < D_; ++d) sv += qhL[tid * D_ + d] * bk[tid * D_ + d];
    qb[tid] = sv;
  }
}

// ---------------------------------------------------------------------------
// k1: h = relu(x @ Wp + bp), fp32 in -> bf16 out, WMMA bf16 64x64 block tiles
// grid: (E/64, B*S/64), block: 256 (8 waves, each owns two 16x16 C tiles)
// ---------------------------------------------------------------------------
__global__ __launch_bounds__(256) void k1_proj_relu_wmma(
    const float* __restrict__ x, const float* __restrict__ Wp,
    const float* __restrict__ bp, __bf16* __restrict__ hOut)
{
  __shared__ __align__(16) __bf16 As[64 * LDA];    // [m][k] row-major
  __shared__ __align__(16) __bf16 Bs[64 * LDA];    // [n][k] (Wp transposed)

  const int tid  = (int)threadIdx.x;
  const int lane = tid & 31;
  const int wave = tid >> 5;
  const int row0 = (int)blockIdx.y * 64;           // flattened (b*s) base
  const int col0 = (int)blockIdx.x * 64;           // e base

  const int mrow  = wave & 3;                      // 4 tile-rows
  const int ncol0 = (wave >> 2) * 2;               // {0,2}: two adjacent tile-cols

  v8f c0 = {}; v8f c1 = {};

  // global-staging assignments
  const int lr = tid >> 2;                         // 64 x-rows, 4 threads/row
  const int lc = (tid & 3) * 8;                    // 8 floats each
  const int wk = tid >> 3;                         // 32 Wp k-rows, 8 threads/row
  const int wc = (tid & 7) * 8;

  // fragment addresses (A 16x32 / B 32x16, both in contiguous k order: k = 16*half + s)
  const int mA    = mrow * 16 + (lane & 15);
  const int ghalf = (lane >> 4);                   // 0: lanes 0-15, 1: lanes 16-31
  const int aoff0 = mA * LDA + ghalf * 16;         // VGPR0-3: K = g*16 .. +7
  const int aoff1 = aoff0 + 8;                     // VGPR4-7: K = g*16+8 .. +7
  const int nB    = lane & 15;
  const int koff  = ghalf * 16;                    // lanes 16-31 hold K=16..31
  const int b0off = (ncol0 * 16 + nB) * LDA + koff;
  const int b1off = b0off + 16 * LDA;

  for (int k0 = 0; k0 < IN_; k0 += 32) {
    // stage x tile (64x32 fp32 -> bf16)
    const float4* xp = (const float4*)(x + (size_t)(row0 + lr) * IN_ + (k0 + lc));
    const float4 xa = xp[0], xb = xp[1];
    v8bf av;
    av[0]=f2bf(xa.x); av[1]=f2bf(xa.y); av[2]=f2bf(xa.z); av[3]=f2bf(xa.w);
    av[4]=f2bf(xb.x); av[5]=f2bf(xb.y); av[6]=f2bf(xb.z); av[7]=f2bf(xb.w);
    *(v8bf*)(As + lr * LDA + lc) = av;

    // stage Wp tile (32x64) transposed into Bs[n][k]
    const float4* wp = (const float4*)(Wp + (size_t)(k0 + wk) * E_ + (col0 + wc));
    const float4 wa = wp[0], wb = wp[1];
    Bs[(wc + 0) * LDA + wk] = f2bf(wa.x);
    Bs[(wc + 1) * LDA + wk] = f2bf(wa.y);
    Bs[(wc + 2) * LDA + wk] = f2bf(wa.z);
    Bs[(wc + 3) * LDA + wk] = f2bf(wa.w);
    Bs[(wc + 4) * LDA + wk] = f2bf(wb.x);
    Bs[(wc + 5) * LDA + wk] = f2bf(wb.y);
    Bs[(wc + 6) * LDA + wk] = f2bf(wb.z);
    Bs[(wc + 7) * LDA + wk] = f2bf(wb.w);
    __syncthreads();

    v16bf a, b0v, b1v;
    {
      const v8bf t0 = *(const v8bf*)(As + aoff0);
      const v8bf t1 = *(const v8bf*)(As + aoff1);
      #pragma unroll
      for (int i = 0; i < 8; ++i) { a[i] = t0[i]; a[i + 8] = t1[i]; }
    }
    {
      const v8bf t0 = *(const v8bf*)(Bs + b0off);
      const v8bf t1 = *(const v8bf*)(Bs + b0off + 8);
      #pragma unroll
      for (int i = 0; i < 8; ++i) { b0v[i] = t0[i]; b0v[i + 8] = t1[i]; }
    }
    {
      const v8bf t0 = *(const v8bf*)(Bs + b1off);
      const v8bf t1 = *(const v8bf*)(Bs + b1off + 8);
      #pragma unroll
      for (int i = 0; i < 8; ++i) { b1v[i] = t0[i]; b1v[i + 8] = t1[i]; }
    }
    c0 = __builtin_amdgcn_wmma_f32_16x16x32_bf16(false, a, false, b0v, (short)0, c0, false, false);
    c1 = __builtin_amdgcn_wmma_f32_16x16x32_bf16(false, a, false, b1v, (short)0, c1, false, false);
    __syncthreads();
  }

  // epilogue: C layout -> element r: M = r + 8*(lane>=16), N = lane&15
  const int gm  = row0 + mrow * 16 + ghalf * 8;
  const int gn0 = col0 + ncol0 * 16 + nB;
  const int gn1 = gn0 + 16;
  const float bpa = bp[gn0];
  const float bpb = bp[gn1];
  #pragma unroll
  for (int r = 0; r < 8; ++r) {
    float v0 = c0[r] + bpa; v0 = v0 > 0.f ? v0 : 0.f;
    float v1 = c1[r] + bpb; v1 = v1 > 0.f ? v1 : 0.f;
    hOut[(size_t)(gm + r) * E_ + gn0] = f2bf(v0);
    hOut[(size_t)(gm + r) * E_ + gn1] = f2bf(v1);
  }
}

// ---------------------------------------------------------------------------
// k2: scoresT[b,s,h] = (h[b,s,:] . qk[h,:] + qb[h]) / 8  (masked)
// ---------------------------------------------------------------------------
__global__ __launch_bounds__(256) void k2_scores(
    const __bf16* __restrict__ h, const float* __restrict__ qk,
    const float* __restrict__ qb, const int* __restrict__ mask,
    float* __restrict__ scoresT)
{
  __shared__ float qkL[H_ * E_];
  const int tid = (int)threadIdx.x;
  for (int i = tid; i < H_ * E_; i += 256) qkL[i] = qk[i];
  __syncthreads();

  const int bs = (int)blockIdx.x * 256 + tid;      // < B*S
  float acc[H_] = {};
  const v8bf* hp = (const v8bf*)(h + (size_t)bs * E_);
  for (int e8 = 0; e8 < E_ / 8; ++e8) {
    const v8bf hv = hp[e8];
    #pragma unroll
    for (int j = 0; j < 8; ++j) {
      const float v = bf2f(hv[j]);
      const int e = e8 * 8 + j;
      #pragma unroll
      for (int hh = 0; hh < H_; ++hh) acc[hh] += v * qkL[hh * E_ + e];
    }
  }
  const int b = bs >> 12, s = bs & (S_ - 1);
  const int mval = mask[b * S_ + s];
  #pragma unroll
  for (int hh = 0; hh < H_; ++hh) {
    float sc = (acc[hh] + qb[hh]) * 0.125f;
    if (mval == 0) sc = -__builtin_inff();
    scoresT[(size_t)bs * H_ + hh] = sc;
  }
}

// ---------------------------------------------------------------------------
// k3: softmax over S for each (b,h), in place on scoresT (B,S,H layout)
// ---------------------------------------------------------------------------
__global__ __launch_bounds__(256) void k3_softmax(float* __restrict__ scoresT)
{
  const int b  = (int)blockIdx.x >> 3;
  const int hh = (int)blockIdx.x & 7;
  const int tid = (int)threadIdx.x;
  __shared__ float red[256];

  float v[16];
  float mx = -__builtin_inff();
  #pragma unroll
  for (int i = 0; i < 16; ++i) {
    const int s = tid + i * 256;
    v[i] = scoresT[((size_t)b * S_ + s) * H_ + hh];
    mx = fmaxf(mx, v[i]);
  }
  red[tid] = mx; __syncthreads();
  for (int off = 128; off > 0; off >>= 1) {
    if (tid < off) red[tid] = fmaxf(red[tid], red[tid + off]);
    __syncthreads();
  }
  mx = red[0]; __syncthreads();

  float sum = 0.f;
  #pragma unroll
  for (int i = 0; i < 16; ++i) { v[i] = __expf(v[i] - mx); sum += v[i]; }
  red[tid] = sum; __syncthreads();
  for (int off = 128; off > 0; off >>= 1) {
    if (tid < off) red[tid] += red[tid + off];
    __syncthreads();
  }
  const float inv = 1.0f / red[0];
  #pragma unroll
  for (int i = 0; i < 16; ++i) {
    const int s = tid + i * 256;
    scoresT[((size_t)b * S_ + s) * H_ + hh] = v[i] * inv;
  }
}

// ---------------------------------------------------------------------------
// k4: ph[b,h,e] = sum_s attn[b,h,s] * h[b,s,e]   grid: (8 e-chunks, B)
// ---------------------------------------------------------------------------
__global__ __launch_bounds__(256) void k4_pool(
    const __bf16* __restrict__ h, const float* __restrict__ attnT,
    float* __restrict__ ph)
{
  const int b  = (int)blockIdx.y;
  const int ec = (int)blockIdx.x;                  // 64-col chunk of E
  const int tid = (int)threadIdx.x;
  const int eg = tid & 15;                         // 16 e-groups of 4
  const int sg = tid >> 4;                         // 16 s-phases
  const int e0 = ec * 64 + eg * 4;

  float acc[H_][4] = {};
  for (int s = sg; s < S_; s += 16) {
    const v4bf hv = *(const v4bf*)(h + ((size_t)b * S_ + s) * E_ + e0);
    float v[4];
    #pragma unroll
    for (int j = 0; j < 4; ++j) v[j] = bf2f(hv[j]);
    const float4* ap = (const float4*)(attnT + ((size_t)b * S_ + s) * H_);
    const float4 a0 = ap[0], a1 = ap[1];
    const float at[H_] = {a0.x, a0.y, a0.z, a0.w, a1.x, a1.y, a1.z, a1.w};
    #pragma unroll
    for (int hh = 0; hh < H_; ++hh)
      #pragma unroll
      for (int j = 0; j < 4; ++j) acc[hh][j] += at[hh] * v[j];
  }

  __shared__ float red[16][16 * 32];               // [sg][eg*32 + h*4+j]
  #pragma unroll
  for (int idx = 0; idx < 32; ++idx) red[sg][eg * 32 + idx] = acc[idx >> 2][idx & 3];
  __syncthreads();
  for (int o = tid; o < 512; o += 256) {
    const int egv = o >> 5, idx = o & 31;
    float sv = 0.f;
    #pragma unroll
    for (int g = 0; g < 16; ++g) sv += red[g][egv * 32 + idx];
    const int hh = idx >> 2, j = idx & 3;
    const int e = ec * 64 + egv * 4 + j;
    ph[((size_t)b * H_ + hh) * E_ + e] = sv;
  }
}

// ---------------------------------------------------------------------------
// k5: pooled = ph@Wv + bv ; out1 = pooled@Wo + bo ; out2 = out1@W2 + b2 ;
//     LayerNorm(out2).   One block per batch row.
// ---------------------------------------------------------------------------
__global__ __launch_bounds__(256) void k5_head(
    const float* __restrict__ ph, const float* __restrict__ Wv,
    const float* __restrict__ bv, const float* __restrict__ Wo,
    const float* __restrict__ bo, const float* __restrict__ W2,
    const float* __restrict__ b2, const float* __restrict__ gamma,
    const float* __restrict__ beta, float* __restrict__ out)
{
  const int b = (int)blockIdx.x;
  const int tid = (int)threadIdx.x;
  __shared__ float phL[H_ * E_];
  __shared__ float buf1[E_];
  __shared__ float buf2[E_];
  __shared__ float redS[256];

  for (int i = tid; i < H_ * E_; i += 256) phL[i] = ph[(size_t)b * H_ * E_ + i];
  __syncthreads();

  for (int o = tid; o < E_; o += 256) {            // pooled (per-head Wv slice)
    const int hh = o >> 6;
    float sv = bv[o];
    const float* phh = phL + hh * E_;
    for (int e = 0; e < E_; ++e) sv += phh[e] * Wv[(size_t)e * E_ + o];
    buf1[o] = sv;
  }
  __syncthreads();
  for (int o = tid; o < E_; o += 256) {            // attn out-proj
    float sv = bo[o];
    for (int i = 0; i < E_; ++i) sv += buf1[i] * Wo[(size_t)i * E_ + o];
    buf2[o] = sv;
  }
  __syncthreads();
  float r[2];
  #pragma unroll
  for (int t = 0; t < 2; ++t) {                    // output projection
    const int o = tid + t * 256;
    float sv = b2[o];
    for (int i = 0; i < E_; ++i) sv += buf2[i] * W2[(size_t)i * E_ + o];
    r[t] = sv;
  }

  // LayerNorm (population variance, eps 1e-5)
  redS[tid] = r[0] + r[1]; __syncthreads();
  for (int off = 128; off > 0; off >>= 1) {
    if (tid < off) redS[tid] += redS[tid + off];
    __syncthreads();
  }
  const float mu = redS[0] * (1.0f / E_); __syncthreads();
  redS[tid] = r[0] * r[0] + r[1] * r[1]; __syncthreads();
  for (int off = 128; off > 0; off >>= 1) {
    if (tid < off) redS[tid] += redS[tid + off];
    __syncthreads();
  }
  const float var = redS[0] * (1.0f / E_) - mu * mu;
  const float inv = rsqrtf(var + 1e-5f);
  #pragma unroll
  for (int t = 0; t < 2; ++t) {
    const int o = tid + t * 256;
    out[(size_t)b * E_ + o] = (r[t] - mu) * inv * gamma[o] + beta[o];
  }
}

// ---------------------------------------------------------------------------
extern "C" void kernel_launch(void* const* d_in, const int* in_sizes, int n_in,
                              void* d_out, int out_size, void* d_ws, size_t ws_size,
                              hipStream_t stream) {
  (void)in_sizes; (void)n_in; (void)out_size; (void)ws_size;
  const float* x     = (const float*)d_in[0];
  const int*   mask  = (const int*)  d_in[1];
  const float* query = (const float*)d_in[2];
  const float* Wp    = (const float*)d_in[3];
  const float* bp    = (const float*)d_in[4];
  const float* Wq    = (const float*)d_in[5];
  const float* bq    = (const float*)d_in[6];
  const float* Wk    = (const float*)d_in[7];
  const float* bk    = (const float*)d_in[8];
  const float* Wv    = (const float*)d_in[9];
  const float* bv    = (const float*)d_in[10];
  const float* Wo    = (const float*)d_in[11];
  const float* bo    = (const float*)d_in[12];
  const float* W2    = (const float*)d_in[13];
  const float* b2    = (const float*)d_in[14];
  const float* gamma = (const float*)d_in[15];
  const float* beta  = (const float*)d_in[16];
  float* out = (float*)d_out;

  char* ws = (char*)d_ws;
  const size_t OFF_H  = 0;                                        // 128 MB bf16 h
  const size_t OFF_SC = OFF_H  + (size_t)B_ * S_ * E_ * 2;        // 4 MB scores/attn
  const size_t OFF_QK = OFF_SC + (size_t)B_ * S_ * H_ * 4;        // 16 KB qk
  const size_t OFF_QB = OFF_QK + (size_t)H_ * E_ * 4;             // qb
  const size_t OFF_PH = OFF_QB + 64;                              // 512 KB ph
  __bf16* hBuf    = (__bf16*)(ws + OFF_H);
  float*  scoresT = (float*) (ws + OFF_SC);
  float*  qk      = (float*) (ws + OFF_QK);
  float*  qb      = (float*) (ws + OFF_QB);
  float*  phBuf   = (float*) (ws + OFF_PH);

  k0_query<<<dim3(1), dim3(256), 0, stream>>>(query, Wq, bq, Wk, bk, qk, qb);
  k1_proj_relu_wmma<<<dim3(E_ / 64, (B_ * S_) / 64), dim3(256), 0, stream>>>(x, Wp, bp, hBuf);
  k2_scores<<<dim3((B_ * S_) / 256), dim3(256), 0, stream>>>(hBuf, qk, qb, mask, scoresT);
  k3_softmax<<<dim3(B_ * H_), dim3(256), 0, stream>>>(scoresT);
  k4_pool<<<dim3(E_ / 64, B_), dim3(256), 0, stream>>>(hBuf, scoresT, phBuf);
  k5_head<<<dim3(B_), dim3(256), 0, stream>>>(phBuf, Wv, bv, Wo, bo, W2, b2, gamma, beta, out);
}
